// LSTMModel_63617055588685
// MI455X (gfx1250) — matrix operand order repaired and checked by reading, not compiled, not run on hardware
//
#include <hip/hip_runtime.h>

// ---------------- problem constants ----------------
#define BATCH   512
#define TSTEPS  2048
#define HID     100
#define GATES   400      // 4*HID
#define KPAD    128      // HID padded: 100 h + 1 x + 1 bias + 26 zero
#define BT      16       // batch rows per workgroup
#define NTILES  25       // GATES/16 column tiles -> one wave each
#define THREADS 800      // 25 waves * 32 lanes
#define NBLOCKS 32       // BATCH/BT

typedef __attribute__((ext_vector_type(16))) __bf16    v16bf;
typedef __attribute__((ext_vector_type(8)))  float     v8f;
typedef __attribute__((ext_vector_type(4)))  unsigned  uv4;

__device__ __forceinline__ unsigned short f2bf(float f) {
    union { float f; unsigned u; } c; c.f = f;
    unsigned u = c.u;
    return (unsigned short)((u + 0x7FFFu + ((u >> 16) & 1u)) >> 16);
}
__device__ __forceinline__ float bf2f(unsigned short s) {
    union { unsigned u; float f; } c; c.u = ((unsigned)s) << 16; return c.f;
}
__device__ __forceinline__ float fsigmoid(float x) {
    // 1/(1+exp(-x)) with v_exp_f32 + v_rcp_f32
    return __builtin_amdgcn_rcpf(1.0f + __builtin_amdgcn_exp2f(-1.44269504f * x));
}
__device__ __forceinline__ float ftanh(float x) {
    // tanh(x) = 1 - 2/(exp(2x)+1); saturates correctly at +/-inf of exp2
    return 1.0f - 2.0f * __builtin_amdgcn_rcpf(1.0f + __builtin_amdgcn_exp2f(2.88539008f * x));
}

union Frag { v16bf v; unsigned short s[16]; uv4 q[2]; };

__global__ __launch_bounds__(THREADS)
void lstm_persistent_kernel(const float* __restrict__ x,      // [B,T,1]
                            const float* __restrict__ W_ih,   // [4H,1]
                            const float* __restrict__ W_hh,   // [4H,H]
                            const float* __restrict__ b_ih,   // [4H]
                            const float* __restrict__ b_hh,   // [4H]
                            const float* __restrict__ W_out,  // [1,H]
                            const float* __restrict__ b_out,  // [1]
                            float* __restrict__ out)          // [B,1]
{
    // h state (bf16, K-padded). row stride 136 (=272B, 16B aligned, conflict-free)
    __shared__ alignas(16) unsigned short hbuf[BT][136];
    // pre-activation gates (f32). row stride 404 -> conflict-free C-layout store
    __shared__ float gbuf[BT][404];

    const int tid  = threadIdx.x;
    const int wave = tid >> 5;        // 0..24 : gate-column tile
    const int lane = tid & 31;
    const int half = lane >> 4;       // wave32 half
    const int lr   = lane & 15;
    const int b0   = blockIdx.x * BT; // first batch row of this WG

    // ---- persistent B fragments: B[k][c] = W_hh[c][k], row100=W_ih, row101=bias ----
    Frag bf[4];
    {
        const int   col  = wave * 16 + lr;                 // 0..399
        const float wih  = W_ih[col];
        const float bias = b_ih[col] + b_hh[col];
        #pragma unroll
        for (int ks = 0; ks < 4; ++ks) {
            #pragma unroll
            for (int j = 0; j < 16; ++j) {
                const int k = ks * 32 + (j < 8 ? half * 8 + j : 16 + half * 8 + (j - 8));     // B-layout: K = 32ks+8*half+j (j<8), 32ks+16+8*half+(j-8) (j>=8)
                float v = 0.0f;
                if (k < HID)           v = W_hh[col * HID + k];
                else if (k == HID)     v = wih;
                else if (k == HID + 1) v = bias;
                bf[ks].s[j] = f2bf(v);
            }
        }
    }

    // ---- init h=0, x slot = x[b,0], ones slot, zero pad ----
    for (int i = tid; i < BT * 136; i += THREADS) {
        const int r = i / 136, k = i % 136;
        unsigned short v = 0;
        if (k == HID)          v = f2bf(x[(size_t)(b0 + r) * TSTEPS]);
        else if (k == HID + 1) v = 0x3F80;                 // bf16 1.0
        hbuf[r][k] = v;
    }

    // ---- per-thread cell state: cells tid and tid+800 of the 1600 (b,j) cells ----
    const int ca = tid,          cb = tid + THREADS;
    const int ba = ca / HID,     ja = ca % HID;
    const int bb = cb / HID,     jb = cb % HID;
    float c_a = 0.0f, c_b = 0.0f;

    __syncthreads();

    for (int t = 0; t < TSTEPS; ++t) {
        // Prefetch next timestep's input scalar early: the global load latency
        // overlaps the GEMM phase instead of sitting between the two barriers.
        float xnext = 0.0f;
        if (tid < BT) {
            int tn = (t + 1 < TSTEPS) ? (t + 1) : t;
            xnext = x[(size_t)(b0 + tid) * TSTEPS + tn];
        }

        // ===== GEMM phase: gates[16 x 16] tile at columns [16*wave, ...) =====
        // Load all 4 A fragments into distinct registers first so the 8
        // ds_load_b128 issue back-to-back and overlap with the WMMA chain.
        Frag af0, af1, af2, af3;
        {
            const unsigned short* rowp = &hbuf[lr][0];     // A-layout: row = lane%16
            const int o = half * 8;
            af0.q[0] = *(const uv4*)(rowp + 0   + o);
            af0.q[1] = *(const uv4*)(rowp + 16  + o);
            af1.q[0] = *(const uv4*)(rowp + 32  + o);
            af1.q[1] = *(const uv4*)(rowp + 48  + o);
            af2.q[0] = *(const uv4*)(rowp + 64  + o);
            af2.q[1] = *(const uv4*)(rowp + 80  + o);
            af3.q[0] = *(const uv4*)(rowp + 96  + o);
            af3.q[1] = *(const uv4*)(rowp + 112 + o);
        }
        v8f acc = {};
        acc = __builtin_amdgcn_wmma_f32_16x16x32_bf16(false, af0.v, false, bf[0].v, (short)0, acc, false, false);
        acc = __builtin_amdgcn_wmma_f32_16x16x32_bf16(false, af1.v, false, bf[1].v, (short)0, acc, false, false);
        acc = __builtin_amdgcn_wmma_f32_16x16x32_bf16(false, af2.v, false, bf[2].v, (short)0, acc, false, false);
        acc = __builtin_amdgcn_wmma_f32_16x16x32_bf16(false, af3.v, false, bf[3].v, (short)0, acc, false, false);

        #pragma unroll
        for (int r = 0; r < 8; ++r)                        // C-layout: M = r + 8*half
            gbuf[half * 8 + r][wave * 16 + lr] = acc[r];
        __syncthreads();

        // ===== elementwise LSTM cell (2 cells/thread), c in registers =====
        {
            const float ig = fsigmoid(gbuf[ba][ja]);
            const float fg = fsigmoid(gbuf[ba][HID + ja]);
            const float gg = ftanh   (gbuf[ba][2 * HID + ja]);
            const float og = fsigmoid(gbuf[ba][3 * HID + ja]);
            c_a = fg * c_a + ig * gg;
            hbuf[ba][ja] = f2bf(og * ftanh(c_a));
        }
        {
            const float ig = fsigmoid(gbuf[bb][jb]);
            const float fg = fsigmoid(gbuf[bb][HID + jb]);
            const float gg = ftanh   (gbuf[bb][2 * HID + jb]);
            const float og = fsigmoid(gbuf[bb][3 * HID + jb]);
            c_b = fg * c_b + ig * gg;
            hbuf[bb][jb] = f2bf(og * ftanh(c_b));
        }
        // stage next timestep's input scalar into the augmented K column
        if (tid < BT)
            hbuf[tid][HID] = f2bf(xnext);
        __syncthreads();
    }

    // ===== readout: out[b] = h_last[b,:] . W_out + b_out  (once, tiny) =====
    if (tid < BT) {
        float s = b_out[0];
        for (int j = 0; j < HID; ++j)
            s += bf2f(hbuf[tid][j]) * W_out[j];
        out[b0 + tid] = s;
    }
}

extern "C" void kernel_launch(void* const* d_in, const int* in_sizes, int n_in,
                              void* d_out, int out_size, void* d_ws, size_t ws_size,
                              hipStream_t stream) {
    (void)in_sizes; (void)n_in; (void)d_ws; (void)ws_size; (void)out_size;
    const float* x     = (const float*)d_in[0];
    const float* W_ih  = (const float*)d_in[1];
    const float* W_hh  = (const float*)d_in[2];
    const float* b_ih  = (const float*)d_in[3];
    const float* b_hh  = (const float*)d_in[4];
    const float* W_out = (const float*)d_in[5];
    const float* b_out = (const float*)d_in[6];
    float* out = (float*)d_out;
    lstm_persistent_kernel<<<dim3(NBLOCKS), dim3(THREADS), 0, stream>>>(
        x, W_ih, W_hh, b_ih, b_hh, W_out, b_out, out);
}
